// Noise_decomp_85718957294136
// MI455X (gfx1250) — matrix operand order repaired and checked by reading, not compiled, not run on hardware
//
#include <hip/hip_runtime.h>
#include <hip/hip_bf16.h>
#include <math.h>

#define BB 16
#define LL 2048
#define DD 512
#define MD 1536   // M*D = 3*512
#define TOPK 7
#define KSTEPS (MD / 32)

typedef __attribute__((ext_vector_type(16))) __bf16 v16bf;
typedef __attribute__((ext_vector_type(8)))  float  v8f;
typedef __attribute__((ext_vector_type(4)))  int    v4i;

struct U32x8 { uint4 lo; uint4 hi; };  // 32 bytes, bit_cast target for v16bf

// gfx1250 async global->LDS path (ASYNCcnt). Guarded: fall back to the
// proven direct-global WMMA loop if the builtins are not declared.
#if defined(__HIP_DEVICE_COMPILE__) && \
    __has_builtin(__builtin_amdgcn_global_load_async_to_lds_b128) && \
    __has_builtin(__builtin_amdgcn_s_wait_asynccnt)
#define HAVE_ASYNC 1
#else
#define HAVE_ASYNC 0
#endif

#if HAVE_ASYNC
typedef __attribute__((address_space(1))) v4i g4i;   // global int4
typedef __attribute__((address_space(3))) v4i l4i;   // LDS int4
#endif

static __device__ __forceinline__ unsigned short f2bf(float f) {
    unsigned int u = __float_as_uint(f);
    u += 0x7FFFu + ((u >> 16) & 1u);   // round-to-nearest-even
    return (unsigned short)(u >> 16);
}

// ---------------- K0: zero the Gram-diagonal accumulator ----------------
__global__ void zero_kernel(float* __restrict__ g, int n) {
    int i = blockIdx.x * blockDim.x + threadIdx.x;
    if (i < n) g[i] = 0.0f;
}

// ---------------- K1: moving averages -> bf16 Y (B,L,1536) and f32 z ----
// ma_k[t] = (1/k) sum_{j=-h..h} x[clamp(t+j)], h=(k-1)/2 (matches edge-repeat pad)
__global__ __launch_bounds__(256)
void decomp_kernel(const float* __restrict__ x,
                   const float* __restrict__ w_out,
                   unsigned short* __restrict__ Ybf,
                   float* __restrict__ z) {
    const int b  = blockIdx.y;
    const int t0 = blockIdx.x * 128;
    const float w0 = w_out[0], w1 = w_out[1], w2 = w_out[2];
    const float* xb = x + (size_t)b * LL * DD;
    unsigned short* Yb = Ybf + (size_t)b * LL * MD;
    float* zb = z + (size_t)b * LL * DD;
    for (int dd = 0; dd < 2; ++dd) {
        const int d = threadIdx.x + dd * 256;
        float s5 = 0.f, s25 = 0.f;
        for (int j = -12; j <= 12; ++j) {
            int tcl = min(max(t0 + j, 0), LL - 1);
            float v = xb[(size_t)tcl * DD + d];
            s25 += v;
            if (j >= -2 && j <= 2) s5 += v;
        }
        for (int t = t0; t < t0 + 128; ++t) {
            float xv  = xb[(size_t)t * DD + d];
            float m5  = s5 * 0.2f;
            float m25 = s25 * (1.0f / 25.0f);
            size_t yrow = (size_t)t * MD;
            Yb[yrow + d]          = f2bf(xv);
            Yb[yrow + DD + d]     = f2bf(m5);
            Yb[yrow + 2 * DD + d] = f2bf(m25);
            zb[(size_t)t * DD + d] = w0 * xv + w1 * m5 + w2 * m25;
            // slide window center t -> t+1
            s5  += xb[(size_t)min(t + 3, LL - 1)  * DD + d] - xb[(size_t)max(t - 2, 0)  * DD + d];
            s25 += xb[(size_t)min(t + 13, LL - 1) * DD + d] - xb[(size_t)max(t - 12, 0) * DD + d];
        }
    }
}

// ---------------- K2: Gram autocorrelation via WMMA bf16 ----------------
// Block = 8 waves over a 128x64 macro-tile of G = Y*Y^T (per batch).
// Async path: per 32-wide K chunk, stage all 192 needed rows (A:128, B:64;
// 64B each, padded to 80B rows in LDS) via GLOBAL_LOAD_ASYNC_TO_LDS_B128,
// double-buffered against the WMMA of the previous chunk. Each wave computes
// a 32x32 region = 2x2 wmma_f32_16x16x32_bf16 tiles, then folds C elements
// into cyclic-diagonal bins (tau = (col-row) mod L) in LDS, then sparse
// global atomics into Gsum[b, tau].

#define SROW 40          // LDS row pitch in ushorts (80B, 16B-aligned, conflict-free)

#if HAVE_ASYNC
static __device__ __forceinline__ void wait_async0() {
    __builtin_amdgcn_s_wait_asynccnt(0);
}
// Stage 192 rows x 32 ushorts: rows 0..127 = A rows I.., rows 128..191 = B rows J..
static __device__ __forceinline__ void stage_tile(const unsigned short* Yb,
                                                  unsigned short* dst,
                                                  int I, int J, int k0, int tid) {
    #pragma unroll
    for (int c = 0; c < 3; ++c) {
        int i   = tid + c * 256;          // 768 16B chunks
        int row = i >> 2;
        int ch  = (i & 3) * 8;            // ushort offset of 16B chunk
        int grow = (row < 128) ? (I + row) : (J + (row - 128));
        const unsigned short* gp = Yb + (size_t)grow * MD + (k0 + ch);
        unsigned short* lp = dst + row * SROW + ch;
        __builtin_amdgcn_global_load_async_to_lds_b128((g4i*)gp, (l4i*)lp, 0, 0);
    }
}
#endif

__global__ __launch_bounds__(256)
void autocorr_wmma_kernel(const unsigned short* __restrict__ Ybf,
                          float* __restrict__ Gsum) {
    __shared__ float bins[256];
    const int b    = blockIdx.z;
    const int I    = blockIdx.y * 128;
    const int J    = blockIdx.x * 64;
    const int tid  = threadIdx.x;
    const int wave = tid >> 5;
    const int lane = tid & 31;
    const int l15  = lane & 15;
    const int lhi  = lane >> 4;
    const int RW = 32 * (wave & 3);   // wave-local row base (within 128)
    const int CW = 32 * (wave >> 2);  // wave-local col base (within 64)
    const unsigned short* Yb = Ybf + (size_t)b * LL * MD;

    if (tid < 256) bins[tid] = 0.0f;

    v8f c00 = {}, c01 = {}, c10 = {}, c11 = {};

    const int ka = 8 * lhi;    // A: elems 0..7 <- k+ka.., elems 8..15 <- k+ka+16..
    const int kb = 8 * lhi;    // B: elems 0..7 <- k+kb.., elems 8..15 <- k+kb+16..

#if HAVE_ASYNC
    __shared__ __align__(16) unsigned short stg[2][192 * SROW];

    stage_tile(Yb, &stg[0][0], I, J, 0, tid);
    wait_async0();
    __syncthreads();

    const int raA0 = (RW + l15) * SROW;
    const int raA1 = (RW + 16 + l15) * SROW;
    const int raB0 = (128 + CW + l15) * SROW;
    const int raB1 = (128 + CW + 16 + l15) * SROW;

    for (int it = 0; it < KSTEPS; ++it) {
        const int p = it & 1;
        if (it + 1 < KSTEPS) stage_tile(Yb, &stg[p ^ 1][0], I, J, (it + 1) * 32, tid);

        const unsigned short* s = &stg[p][0];
        U32x8 pa0, pa1, pb0, pb1;
        pa0.lo = *(const uint4*)(s + raA0 + ka);
        pa0.hi = *(const uint4*)(s + raA0 + ka + 16);
        pa1.lo = *(const uint4*)(s + raA1 + ka);
        pa1.hi = *(const uint4*)(s + raA1 + ka + 16);
        pb0.lo = *(const uint4*)(s + raB0 + kb);
        pb0.hi = *(const uint4*)(s + raB0 + kb + 16);
        pb1.lo = *(const uint4*)(s + raB1 + kb);
        pb1.hi = *(const uint4*)(s + raB1 + kb + 16);
        v16bf a0 = __builtin_bit_cast(v16bf, pa0);
        v16bf a1 = __builtin_bit_cast(v16bf, pa1);
        v16bf b0 = __builtin_bit_cast(v16bf, pb0);
        v16bf b1 = __builtin_bit_cast(v16bf, pb1);
        c00 = __builtin_amdgcn_wmma_f32_16x16x32_bf16(false, a0, false, b0, (short)0, c00, false, false);
        c01 = __builtin_amdgcn_wmma_f32_16x16x32_bf16(false, a0, false, b1, (short)0, c01, false, false);
        c10 = __builtin_amdgcn_wmma_f32_16x16x32_bf16(false, a1, false, b0, (short)0, c10, false, false);
        c11 = __builtin_amdgcn_wmma_f32_16x16x32_bf16(false, a1, false, b1, (short)0, c11, false, false);

        __syncthreads();   // all ds reads of stg[p] complete
        wait_async0();     // this wave's async writes to stg[p^1] landed
        __syncthreads();   // visible to all waves before next iteration
    }
#else
    __syncthreads();
    // Fallback: direct global loads (round-1 proven codegen)
    const unsigned short* ar0 = Yb + (size_t)(I + RW + l15) * MD;
    const unsigned short* ar1 = Yb + (size_t)(I + RW + 16 + l15) * MD;
    const unsigned short* br0 = Yb + (size_t)(J + CW + l15) * MD;
    const unsigned short* br1 = Yb + (size_t)(J + CW + 16 + l15) * MD;
    for (int k0 = 0; k0 < MD; k0 += 32) {
        U32x8 pa0, pa1, pb0, pb1;
        pa0.lo = *(const uint4*)(ar0 + k0 + ka);
        pa0.hi = *(const uint4*)(ar0 + k0 + ka + 16);
        pa1.lo = *(const uint4*)(ar1 + k0 + ka);
        pa1.hi = *(const uint4*)(ar1 + k0 + ka + 16);
        pb0.lo = *(const uint4*)(br0 + k0 + kb);
        pb0.hi = *(const uint4*)(br0 + k0 + kb + 16);
        pb1.lo = *(const uint4*)(br1 + k0 + kb);
        pb1.hi = *(const uint4*)(br1 + k0 + kb + 16);
        v16bf a0 = __builtin_bit_cast(v16bf, pa0);
        v16bf a1 = __builtin_bit_cast(v16bf, pa1);
        v16bf b0 = __builtin_bit_cast(v16bf, pb0);
        v16bf b1 = __builtin_bit_cast(v16bf, pb1);
        c00 = __builtin_amdgcn_wmma_f32_16x16x32_bf16(false, a0, false, b0, (short)0, c00, false, false);
        c01 = __builtin_amdgcn_wmma_f32_16x16x32_bf16(false, a0, false, b1, (short)0, c01, false, false);
        c10 = __builtin_amdgcn_wmma_f32_16x16x32_bf16(false, a1, false, b0, (short)0, c10, false, false);
        c11 = __builtin_amdgcn_wmma_f32_16x16x32_bf16(false, a1, false, b1, (short)0, c11, false, false);
    }
#endif

    // C/D layout: VGPR v holds row m=v (lanes 0-15) or m=v+8 (lanes 16-31); col n=lane&15
    // bin = (col - row) - (J - I - 127) = 127 + (local_col) - (local_row)
    const int rb = RW + 8 * lhi;
    const int cb = CW + l15;
    #pragma unroll
    for (int v = 0; v < 8; ++v) {
        int r0 = rb + v, r1 = r0 + 16;
        atomicAdd(&bins[127 + cb      - r0], c00[v]);
        atomicAdd(&bins[127 + cb + 16 - r0], c01[v]);
        atomicAdd(&bins[127 + cb      - r1], c10[v]);
        atomicAdd(&bins[127 + cb + 16 - r1], c11[v]);
    }
    __syncthreads();
    const int base = J - I - 127;
    for (int i = tid; i < 191; i += 256) {
        int tau = (base + i) & (LL - 1);
        atomicAdd(&Gsum[b * LL + tau], bins[i]);
    }
}

// ---------------- K3: batch-mean top-7 lags + per-batch softmax ---------
__global__ __launch_bounds__(256)
void topk_softmax_kernel(const float* __restrict__ Gsum,
                         int* __restrict__ idx_out,
                         float* __restrict__ tc_out) {
    __shared__ float avg[LL];
    __shared__ float redv[256];
    __shared__ int   redi[256];
    __shared__ int   sidx[TOPK];
    const int tid = threadIdx.x;
    for (int t = tid; t < LL; t += 256) {
        float s = 0.f;
        for (int b = 0; b < BB; ++b) s += Gsum[b * LL + t];
        avg[t] = s;                               // positive scale: ordering unchanged
    }
    __syncthreads();
    for (int it = 0; it < TOPK; ++it) {
        float best = -__builtin_inff(); int bi = 0;
        for (int t = tid; t < LL; t += 256) {
            float v = avg[t];
            if (v > best) { best = v; bi = t; }
        }
        redv[tid] = best; redi[tid] = bi;
        __syncthreads();
        for (int s = 128; s > 0; s >>= 1) {
            if (tid < s) {
                bool take = (redv[tid + s] > redv[tid]) ||
                            (redv[tid + s] == redv[tid] && redi[tid + s] < redi[tid]);
                if (take) { redv[tid] = redv[tid + s]; redi[tid] = redi[tid + s]; }
            }
            __syncthreads();
        }
        if (tid == 0) {
            sidx[it] = redi[0];
            idx_out[it] = redi[0];
            avg[redi[0]] = -__builtin_inff();
        }
        __syncthreads();
    }
    if (tid < BB) {
        float w[TOPK]; float mx = -__builtin_inff();
        for (int i = 0; i < TOPK; ++i) {
            w[i] = Gsum[tid * LL + sidx[i]] * (1.0f / (float)MD);  // corr.mean over (m,d)
            mx = fmaxf(mx, w[i]);
        }
        float se = 0.f;
        for (int i = 0; i < TOPK; ++i) { w[i] = __expf(w[i] - mx); se += w[i]; }
        float inv = 1.0f / se;
        for (int i = 0; i < TOPK; ++i) tc_out[tid * TOPK + i] = w[i] * inv;
    }
}

// ---------------- K4: 7-tap circular gather, res + moving_mean ----------
__global__ __launch_bounds__(128)
void out_kernel(const float* __restrict__ x,
                const float* __restrict__ z,
                const int* __restrict__ idxv,
                const float* __restrict__ tc,
                const float* __restrict__ b_out,
                float* __restrict__ out) {
    const int t = blockIdx.x;
    const int b = blockIdx.y;
    const int d = threadIdx.x * 4;
    const float bias = b_out[0];
    float4 acc = make_float4(bias, bias, bias, bias);
    const float* zb = z + (size_t)b * LL * DD;
    #pragma unroll
    for (int i = 0; i < TOPK; ++i) {
        int ts = (t + idxv[i]) & (LL - 1);
        float w = tc[b * TOPK + i];
        const float4 zv = *(const float4*)(zb + (size_t)ts * DD + d);
        acc.x += w * zv.x; acc.y += w * zv.y; acc.z += w * zv.z; acc.w += w * zv.w;
    }
    const size_t off = ((size_t)b * LL + t) * DD + d;
    const float4 xv = *(const float4*)(x + off);
    float4 res = make_float4(xv.x - acc.x, xv.y - acc.y, xv.z - acc.z, xv.w - acc.w);
    *(float4*)(out + off) = res;                              // res = x - out
    *(float4*)(out + (size_t)BB * LL * DD + off) = acc;       // moving_mean = out
}

extern "C" void kernel_launch(void* const* d_in, const int* in_sizes, int n_in,
                              void* d_out, int out_size, void* d_ws, size_t ws_size,
                              hipStream_t stream) {
    const float* x     = (const float*)d_in[0];
    const float* w_out = (const float*)d_in[1];
    const float* b_out = (const float*)d_in[2];

    char* ws = (char*)d_ws;
    const size_t szY = (size_t)BB * LL * MD * sizeof(unsigned short); // ~96 MB
    const size_t szZ = (size_t)BB * LL * DD * sizeof(float);          // 64 MB
    const size_t szG = (size_t)BB * LL * sizeof(float);               // 128 KB
    size_t offY = 0;
    size_t offZ = (offY + szY + 255) & ~(size_t)255;
    size_t offG = (offZ + szZ + 255) & ~(size_t)255;
    size_t offI = (offG + szG + 255) & ~(size_t)255;
    size_t offT = offI + 256;

    unsigned short* Ybf = (unsigned short*)(ws + offY);
    float* z    = (float*)(ws + offZ);
    float* Gsum = (float*)(ws + offG);
    int*   idxv = (int*)(ws + offI);
    float* tc   = (float*)(ws + offT);

    zero_kernel<<<(BB * LL + 255) / 256, 256, 0, stream>>>(Gsum, BB * LL);
    decomp_kernel<<<dim3(LL / 128, BB), 256, 0, stream>>>(x, w_out, Ybf, z);
    autocorr_wmma_kernel<<<dim3(LL / 64, LL / 128, BB), 256, 0, stream>>>(Ybf, Gsum);
    topk_softmax_kernel<<<1, 256, 0, stream>>>(Gsum, idxv, tc);
    out_kernel<<<dim3(LL, BB), 128, 0, stream>>>(x, z, idxv, tc, b_out, (float*)d_out);
}
